// MolecularGCN_1563368095867
// MI455X (gfx1250) — matrix operand order repaired and checked by reading, not compiled, not run on hardware
//
#include <hip/hip_runtime.h>
#include <hip/hip_bf16.h>

// ---------------- problem constants (match reference) ----------------
#define N_NODES 100000
#define N_EDGES 1200000
#define FDIM    128
#define DDIM    64
#define GDIM    2048
#define LNUM    3
#define EPSV    1e-5f

typedef __attribute__((ext_vector_type(16))) _Float16 v16h;
typedef __attribute__((ext_vector_type(2)))  _Float16 h2;
typedef __attribute__((ext_vector_type(8)))  float    v8f;

// ---------------------------------------------------------------------
// small elementwise / init kernels
// ---------------------------------------------------------------------
__global__ void fill_kernel(float* __restrict__ p, float v, int n) {
    int i = blockIdx.x * blockDim.x + threadIdx.x;
    if (i < n) p[i] = v;
}

__global__ void deg_kernel(const int* __restrict__ dst, float* __restrict__ deg, int e) {
    int i = blockIdx.x * blockDim.x + threadIdx.x;
    if (i < e) atomicAdd(&deg[dst[i]], 1.0f);
}

__global__ void dinv_kernel(const float* __restrict__ deg, float* __restrict__ dinv, int n) {
    int i = blockIdx.x * blockDim.x + threadIdx.x;
    if (i < n) dinv[i] = 1.0f / deg[i];
}

__global__ void norm_kernel(const int* __restrict__ src, const int* __restrict__ dst,
                            const float* __restrict__ deg, float* __restrict__ norm, int e) {
    int i = blockIdx.x * blockDim.x + threadIdx.x;
    if (i < e) norm[i] = rsqrtf(deg[src[i]]) * rsqrtf(deg[dst[i]]);
}

__global__ void relu_kernel(float* __restrict__ x, int n) {
    int i = blockIdx.x * blockDim.x + threadIdx.x;
    if (i < n) x[i] = fmaxf(x[i], 0.0f);
}

// agg init with self-loop term: agg[n,c] = t[n,c] * dinv[n] + b[c]
__global__ void self_init_kernel(const float* __restrict__ t, const float* __restrict__ dinv,
                                 const float* __restrict__ b, float* __restrict__ agg, int n) {
    int i = blockIdx.x * blockDim.x + threadIdx.x;
    int total = n * DDIM;
    if (i < total) {
        int node = i >> 6;
        int c    = i & 63;
        agg[i] = t[i] * dinv[node] + b[c];
    }
}

// edge scatter: agg[dst] += t[src] * norm  (4 features per thread, float4 gather)
__global__ void scatter_kernel(const int* __restrict__ src, const int* __restrict__ dst,
                               const float* __restrict__ norm, const float* __restrict__ t,
                               float* __restrict__ agg, int e) {
    int i = blockIdx.x * blockDim.x + threadIdx.x;
    int total = e * 16;                      // 16 chunks of 4 features per edge
    if (i >= total) return;
    int ed = i >> 4;
    int c  = (i & 15) << 2;
    float nrm = norm[ed];
    const float4 hv = *(const float4*)(t + (size_t)src[ed] * DDIM + c);
    float* o = agg + (size_t)dst[ed] * DDIM + c;
    atomicAdd(o + 0, hv.x * nrm);
    atomicAdd(o + 1, hv.y * nrm);
    atomicAdd(o + 2, hv.z * nrm);
    atomicAdd(o + 3, hv.w * nrm);
}

// global_add_pool: p[batch[n]] += h[n]   (4 features per thread)
__global__ void pool_kernel(const int* __restrict__ batch, const float* __restrict__ h,
                            float* __restrict__ p, int n) {
    int i = blockIdx.x * blockDim.x + threadIdx.x;
    int total = n * 16;
    if (i >= total) return;
    int node = i >> 4;
    int c    = (i & 15) << 2;
    const float4 hv = *(const float4*)(h + (size_t)node * DDIM + c);
    float* o = p + (size_t)batch[node] * DDIM + c;
    atomicAdd(o + 0, hv.x);
    atomicAdd(o + 1, hv.y);
    atomicAdd(o + 2, hv.z);
    atomicAdd(o + 3, hv.w);
}

// ---------------------------------------------------------------------
// BatchNorm (training mode, biased variance) — two pass
// ---------------------------------------------------------------------
__global__ void bn_stats_kernel(const float* __restrict__ x, float* __restrict__ stats, int M) {
    int col    = threadIdx.x & 63;
    int rlane  = threadIdx.x >> 6;            // 0..3
    int row    = blockIdx.x * 4 + rlane;
    int stride = gridDim.x * 4;
    float s = 0.0f, s2 = 0.0f;
    for (; row < M; row += stride) {
        float v = x[(size_t)row * DDIM + col];
        s  += v;
        s2 += v * v;
    }
    atomicAdd(&stats[col],      s);
    atomicAdd(&stats[64 + col], s2);
}

__global__ void bn_apply_kernel(const float* __restrict__ x, const float* __restrict__ stats,
                                const float* __restrict__ g, const float* __restrict__ be,
                                float* __restrict__ y, int M) {
    int i = blockIdx.x * blockDim.x + threadIdx.x;
    int total = M * DDIM;
    if (i >= total) return;
    int c = i & 63;
    float inv = 1.0f / (float)M;
    float mu  = stats[c] * inv;
    float var = stats[64 + c] * inv - mu * mu;
    y[i] = (x[i] - mu) * rsqrtf(var + EPSV) * g[c] + be[c];
}

// ---------------------------------------------------------------------
// WMMA GEMM: C[M x 64] = A[M x K] @ W[K x 64]  (+ optional bias)
// K compile-time (64 or 128) -> fully unrolled K loop.
// Block = 128 threads = 4 wave32; block owns a 32-row strip.
// A tile (32xK) and W (Kx64) staged ONCE in LDS as packed f16
// (conversion out of the hot loop); wave w owns column tile
// [16w,16w+16) and computes two 16x16 tiles sharing one B fragment.
// f32 accumulate via v_wmma_f32_16x16x32_f16.
// ---------------------------------------------------------------------
template<int K>
__global__ __launch_bounds__(128)
void gemm_wmma(const float* __restrict__ A, const float* __restrict__ W,
               const float* __restrict__ bias, float* __restrict__ C, int M) {
    constexpr int KH = K / 2;
    __shared__ __align__(16) _Float16 sA[32 * K];   // [row][k]   row-major halves
    __shared__ __align__(16) _Float16 sB[64 * K];   // [col][k]   per-column packed

    const int tid = threadIdx.x;
    const size_t rowbase = (size_t)blockIdx.x * 32;

    // prefetch next block's A strip into L2 (global_prefetch_b8 path)
    if (tid < 32) {
        const float* nxt = A + (rowbase + 32 + tid) * K;
        __builtin_prefetch(nxt, 0, 1);
    }

    // ---- stage A tile: 32 x K f32 -> packed f16 (row-major) ----
    for (int i = tid; i < 32 * KH; i += 128) {
        int r  = i / KH;
        int kp = i - r * KH;
        float2 f = *(const float2*)(A + (rowbase + r) * K + 2 * kp);
        sA[r * K + 2 * kp]     = (_Float16)f.x;
        sA[r * K + 2 * kp + 1] = (_Float16)f.y;
    }
    // ---- stage W: [k][col] f32 -> sB[col][k] f16 ----
    for (int i = tid; i < 64 * KH; i += 128) {
        int col = i / KH;
        int kp  = i - col * KH;
        sB[col * K + 2 * kp]     = (_Float16)W[(2 * kp)     * DDIM + col];
        sB[col * K + 2 * kp + 1] = (_Float16)W[(2 * kp + 1) * DDIM + col];
    }
    __syncthreads();

    const int lane = tid & 31;
    const int wave = tid >> 5;
    const int mrow = lane & 15;
    const int hi   = lane >> 4;                     // lane half (0/1)
    const int col  = wave * 16 + mrow;              // output column (0..63)

    v8f acc0 = {};                                  // rows rowbase+0..15
    v8f acc1 = {};                                  // rows rowbase+16..31

    #pragma unroll
    for (int kk = 0; kk < K; kk += 32) {
        v16h a0, a1, b;
        #pragma unroll
        for (int v = 0; v < 8; ++v) {
            // A-fragment ISA layout: K0 = 16*(v>=4) + 8*(lane>=16) + 2*(v%4)
            int k0 = kk + ((v & 4) ? 16 : 0) + (hi ? 8 : 0) + 2 * (v & 3);
            h2 pa0 = *(const h2*)(&sA[mrow * K + k0]);
            h2 pa1 = *(const h2*)(&sA[(mrow + 16) * K + k0]);
            a0[2 * v] = pa0.x;  a0[2 * v + 1] = pa0.y;
            a1[2 * v] = pa1.x;  a1[2 * v + 1] = pa1.y;
            // B-fragment: col = lane%16, same K layout as A: K0 = 16*(v>=4) + 8*(lane>=16) + 2*(v%4)
            int kb = kk + ((v & 4) ? 16 : 0) + (hi ? 8 : 0) + 2 * (v & 3);
            h2 pb = *(const h2*)(&sB[col * K + kb]);
            b[2 * v] = pb.x;  b[2 * v + 1] = pb.y;
        }
        acc0 = __builtin_amdgcn_wmma_f32_16x16x32_f16(
            false, a0, false, b, (short)0, acc0, false, false);
        acc1 = __builtin_amdgcn_wmma_f32_16x16x32_f16(
            false, a1, false, b, (short)0, acc1, false, false);
    }

    const float bv = bias ? bias[col] : 0.0f;
    #pragma unroll
    for (int r = 0; r < 8; ++r) {
        int row0 = (int)rowbase + r + (hi ? 8 : 0);          // C/D layout: M = r + 8*hi
        C[(size_t)row0 * DDIM + col] = acc0[r] + bv;
    }
    #pragma unroll
    for (int r = 0; r < 8; ++r) {
        int row1 = (int)rowbase + 16 + r + (hi ? 8 : 0);
        C[(size_t)row1 * DDIM + col] = acc1[r] + bv;
    }
}

// final projection: out[g] = dot(y[g,:], Wo) + bo
__global__ void out_kernel(const float* __restrict__ y, const float* __restrict__ Wo,
                           const float* __restrict__ bo, float* __restrict__ out, int Gn) {
    int g = blockIdx.x * blockDim.x + threadIdx.x;
    if (g >= Gn) return;
    float s = bo[0];
    const float* yr = y + (size_t)g * DDIM;
    #pragma unroll
    for (int c = 0; c < DDIM; ++c) s += yr[c] * Wo[c];
    out[g] = s;
}

// ---------------------------------------------------------------------
// host orchestration
// ---------------------------------------------------------------------
static inline int cdiv(int a, int b) { return (a + b - 1) / b; }

extern "C" void kernel_launch(void* const* d_in, const int* in_sizes, int n_in,
                              void* d_out, int out_size, void* d_ws, size_t ws_size,
                              hipStream_t stream) {
    (void)in_sizes; (void)n_in; (void)out_size; (void)ws_size;

    // inputs in setup_inputs() order
    const float* x    = (const float*)d_in[0];
    const int*   src  = (const int*)d_in[1];
    const int*   dst  = src + N_EDGES;            // edge_index[1]
    const int*   bat  = (const int*)d_in[2];
    const float* W1   = (const float*)d_in[3];
    const float* b1   = (const float*)d_in[4];
    const float* g1   = (const float*)d_in[5];
    const float* be1  = (const float*)d_in[6];
    const float* Wh   = (const float*)d_in[7];    // [L,64,64]
    const float* bh   = (const float*)d_in[8];
    const float* gh   = (const float*)d_in[9];
    const float* beh  = (const float*)d_in[10];
    const float* Wm   = (const float*)d_in[11];
    const float* bm   = (const float*)d_in[12];
    const float* g2   = (const float*)d_in[13];
    const float* be2  = (const float*)d_in[14];
    const float* Wd   = (const float*)d_in[15];   // [L,64,64]
    const float* bd   = (const float*)d_in[16];
    const float* gd   = (const float*)d_in[17];
    const float* bed  = (const float*)d_in[18];
    const float* Wo   = (const float*)d_in[19];
    const float* bo   = (const float*)d_in[20];
    float* out = (float*)d_out;

    // workspace carve-up (256B aligned)
    char*  base = (char*)d_ws;
    size_t off  = 0;
    auto carve = [&](size_t bytes) -> float* {
        float* r = (float*)(base + off);
        off = (off + bytes + 255) & ~(size_t)255;
        return r;
    };
    float* deg   = carve((size_t)N_NODES * 4);
    float* dinv  = carve((size_t)N_NODES * 4);
    float* enorm = carve((size_t)N_EDGES * 4);
    // pad node buffers to a multiple of 32 rows so the 32-row GEMM tile and
    // its L2 prefetch stay in-bounds (N_NODES is already a multiple of 32).
    float* bufT  = carve(((size_t)N_NODES + 32) * DDIM * 4);  // h @ W
    float* bufAg = carve((size_t)N_NODES * DDIM * 4);         // aggregated
    float* bufH  = carve(((size_t)N_NODES + 32) * DDIM * 4);  // layer output
    float* pbuf  = carve(((size_t)GDIM + 32) * DDIM * 4);
    float* yA    = carve(((size_t)GDIM + 32) * DDIM * 4);
    float* yB    = carve(((size_t)GDIM + 32) * DDIM * 4);
    float* stats = carve(128 * 4);

    const int T = 256;
    const int nfeat  = N_NODES * DDIM;
    const int gfeat  = GDIM * DDIM;
    const int e16    = N_EDGES * 16;
    const int n16    = N_NODES * 16;

    // --- degree / norm (self-loop: deg starts at 1) ---
    fill_kernel<<<cdiv(N_NODES, T), T, 0, stream>>>(deg, 1.0f, N_NODES);
    deg_kernel <<<cdiv(N_EDGES, T), T, 0, stream>>>(dst, deg, N_EDGES);
    dinv_kernel<<<cdiv(N_NODES, T), T, 0, stream>>>(deg, dinv, N_NODES);
    norm_kernel<<<cdiv(N_EDGES, T), T, 0, stream>>>(src, dst, deg, enorm, N_EDGES);

    // --- conv layer 1: h = bn(relu(agg(x@W1) + (x@W1)*dinv + b1)) ---
    gemm_wmma<FDIM><<<N_NODES / 32, 128, 0, stream>>>(x, W1, nullptr, bufT, N_NODES);
    self_init_kernel<<<cdiv(nfeat, T), T, 0, stream>>>(bufT, dinv, b1, bufAg, N_NODES);
    scatter_kernel<<<cdiv(e16, T), T, 0, stream>>>(src, dst, enorm, bufT, bufAg, N_EDGES);
    relu_kernel<<<cdiv(nfeat, T), T, 0, stream>>>(bufAg, nfeat);
    fill_kernel<<<1, 128, 0, stream>>>(stats, 0.0f, 128);
    bn_stats_kernel<<<512, 256, 0, stream>>>(bufAg, stats, N_NODES);
    bn_apply_kernel<<<cdiv(nfeat, T), T, 0, stream>>>(bufAg, stats, g1, be1, bufH, N_NODES);

    // --- hidden conv layers (no relu) ---
    for (int i = 0; i < LNUM; ++i) {
        gemm_wmma<DDIM><<<N_NODES / 32, 128, 0, stream>>>(bufH, Wh + (size_t)i * DDIM * DDIM,
                                                          nullptr, bufT, N_NODES);
        self_init_kernel<<<cdiv(nfeat, T), T, 0, stream>>>(bufT, dinv, bh + i * DDIM, bufAg, N_NODES);
        scatter_kernel<<<cdiv(e16, T), T, 0, stream>>>(src, dst, enorm, bufT, bufAg, N_EDGES);
        fill_kernel<<<1, 128, 0, stream>>>(stats, 0.0f, 128);
        bn_stats_kernel<<<512, 256, 0, stream>>>(bufAg, stats, N_NODES);
        bn_apply_kernel<<<cdiv(nfeat, T), T, 0, stream>>>(bufAg, stats, gh + i * DDIM,
                                                          beh + i * DDIM, bufH, N_NODES);
    }

    // --- global_add_pool ---
    fill_kernel<<<cdiv(gfeat, T), T, 0, stream>>>(pbuf, 0.0f, gfeat);
    pool_kernel<<<cdiv(n16, T), T, 0, stream>>>(bat, bufH, pbuf, N_NODES);

    // --- dense head: y = bn(relu(p@Wm+bm)) ---
    gemm_wmma<DDIM><<<GDIM / 32, 128, 0, stream>>>(pbuf, Wm, bm, yA, GDIM);
    relu_kernel<<<cdiv(gfeat, T), T, 0, stream>>>(yA, gfeat);
    fill_kernel<<<1, 128, 0, stream>>>(stats, 0.0f, 128);
    bn_stats_kernel<<<128, 256, 0, stream>>>(yA, stats, GDIM);
    bn_apply_kernel<<<cdiv(gfeat, T), T, 0, stream>>>(yA, stats, g2, be2, yB, GDIM);

    for (int i = 0; i < LNUM; ++i) {
        gemm_wmma<DDIM><<<GDIM / 32, 128, 0, stream>>>(yB, Wd + (size_t)i * DDIM * DDIM,
                                                       bd + i * DDIM, yA, GDIM);
        relu_kernel<<<cdiv(gfeat, T), T, 0, stream>>>(yA, gfeat);
        fill_kernel<<<1, 128, 0, stream>>>(stats, 0.0f, 128);
        bn_stats_kernel<<<128, 256, 0, stream>>>(yA, stats, GDIM);
        bn_apply_kernel<<<cdiv(gfeat, T), T, 0, stream>>>(yA, stats, gd + i * DDIM,
                                                          bed + i * DDIM, yB, GDIM);
    }

    // --- output projection ---
    out_kernel<<<cdiv(GDIM, T), T, 0, stream>>>(yB, Wo, bo, out, GDIM);
}
